// HGAT_65274912964677
// MI455X (gfx1250) — matrix operand order repaired and checked by reading, not compiled, not run on hardware
//
#include <hip/hip_runtime.h>
#include <cstddef>

// ---------------------------------------------------------------------------
// Types for CDNA5 WMMA (wave32)
// ---------------------------------------------------------------------------
typedef __attribute__((ext_vector_type(16))) _Float16 v16h;
typedef __attribute__((ext_vector_type(8)))  _Float16 v8h;
typedef __attribute__((ext_vector_type(8)))  float    v8f;

#define NEGV (-1e30f)

// float atomic-max via signed/unsigned int trick (order-preserving encodings)
__device__ __forceinline__ void atomicMaxF(float* addr, float v) {
    if (v >= 0.0f) atomicMax((int*)addr, __float_as_int(v));
    else           atomicMin((unsigned int*)addr, __float_as_uint(v));
}

__device__ __forceinline__ v8h cvt8(float4 a, float4 b) {
    v8h r;
    r[0] = (_Float16)a.x; r[1] = (_Float16)a.y;
    r[2] = (_Float16)a.z; r[3] = (_Float16)a.w;
    r[4] = (_Float16)b.x; r[5] = (_Float16)b.y;
    r[6] = (_Float16)b.z; r[7] = (_Float16)b.w;
    return r;
}

// ---------------------------------------------------------------------------
// Generic WMMA GEMM: C[M,Ncols] = act( A[M,K] @ B[K,Ncols] + bias )
//   - A,B f32 in global; converted to f16 while staging to LDS (f32 accumulate)
//   - block tile 128(M) x 64(N), BK=32, 8 waves (256 threads)
//   - staging: thread owns contiguous row segments -> vector loads/stores,
//     only a row-level guard on the single ragged M-edge block
//   - Requires Ncols % 64 == 0 and K % 32 == 0 (true for all call sites)
//   - act: 0 = none, 1 = ELU
// ---------------------------------------------------------------------------
__global__ __launch_bounds__(256) void k_gemm(
    const float* __restrict__ A, const float* __restrict__ B,
    const float* __restrict__ bias, float* __restrict__ C,
    int M, int Ncols, int K, int lda, int ldb, int ldc, int act)
{
    // row stride 40 halfwords = 80B -> every row base is 16B aligned
    __shared__ __align__(16) _Float16 As[128 * 40];     // [m][k]
    __shared__ __align__(16) _Float16 Bs[64 * 40];      // [n][k] (transposed)

    const int t    = threadIdx.x;
    const int wave = t >> 5;
    const int lane = t & 31;
    const int g    = lane >> 4;       // half-wave select
    const int l16  = lane & 15;
    const int m0   = blockIdx.x * 128;
    const int n0   = blockIdx.y * 64;

    // staging ownership (contiguous segments; all 16B aligned)
    const int ar = t >> 1;            // A row 0..127
    const int ac = (t & 1) * 16;      // A col base {0,16}
    const int bk = t >> 3;            // B k-row 0..31
    const int bc = (t & 7) * 8;       // B col base 0..56

    const bool fullM  = (m0 + 128 <= M);
    const bool arowOK = fullM || (m0 + ar < M);

    v8f acc[4] = {};                   // 4 x (16x16 f32) accumulators

    const float* pA = A + (size_t)(m0 + ar) * lda + ac;
    const float* pB = B + (size_t)bk * ldb + n0 + bc;

    for (int kk = 0; kk < K; kk += 32) {
        // ---- stage A tile: 16 contiguous f32 per thread -> 2x b128 LDS ----
        float4 a0, a1, a2, a3;
        if (arowOK) {
            const float4* p = (const float4*)(pA + kk);
            a0 = p[0]; a1 = p[1]; a2 = p[2]; a3 = p[3];
        } else {
            a0 = a1 = a2 = a3 = make_float4(0.f, 0.f, 0.f, 0.f);
        }
        *(v8h*)&As[ar * 40 + ac]     = cvt8(a0, a1);
        *(v8h*)&As[ar * 40 + ac + 8] = cvt8(a2, a3);

        // ---- stage B tile: 8 contiguous f32 per thread, transposed store ----
        {
            const float4* p = (const float4*)(pB + (size_t)kk * ldb);
            float4 b0 = p[0], b1 = p[1];
            Bs[(bc + 0) * 40 + bk] = (_Float16)b0.x;
            Bs[(bc + 1) * 40 + bk] = (_Float16)b0.y;
            Bs[(bc + 2) * 40 + bk] = (_Float16)b0.z;
            Bs[(bc + 3) * 40 + bk] = (_Float16)b0.w;
            Bs[(bc + 4) * 40 + bk] = (_Float16)b1.x;
            Bs[(bc + 5) * 40 + bk] = (_Float16)b1.y;
            Bs[(bc + 6) * 40 + bk] = (_Float16)b1.z;
            Bs[(bc + 7) * 40 + bk] = (_Float16)b1.w;
        }
        // prefetch next K-tile into cache while we compute (global_prefetch_b8)
        if (kk + 32 < K) {
            if (arowOK) __builtin_prefetch(pA + kk + 32, 0, 0);
            __builtin_prefetch(pB + (size_t)(kk + 32) * ldb, 0, 0);
        }
        __syncthreads();

        // ---- A fragment (16x32 f16): lane l16 = row; per ISA layout
        //      elems 0..7 -> K = 8g..8g+7 ; elems 8..15 -> K = 16+8g..16+8g+7
        {
            const int arow = wave * 16 + l16;
            v8h alo = *(const v8h*)&As[arow * 40 + g * 8];
            v8h ahi = *(const v8h*)&As[arow * 40 + 16 + g * 8];
            v16h a;
            #pragma unroll
            for (int j = 0; j < 8; ++j) { a[j] = alo[j]; a[j + 8] = ahi[j]; }

            #pragma unroll
            for (int nt = 0; nt < 4; ++nt) {
                // B fragment (32x16 f16): lane l16 = col n; elems 0..7 -> K = 8g+j ; elems 8..15 -> K = 16+8g+j
                const int bn = nt * 16 + l16;
                v8h blo = *(const v8h*)&Bs[bn * 40 + g * 8];
                v8h bhi = *(const v8h*)&Bs[bn * 40 + 16 + g * 8];
                v16h b;
                #pragma unroll
                for (int j = 0; j < 8; ++j) { b[j] = blo[j]; b[j + 8] = bhi[j]; }

                acc[nt] = __builtin_amdgcn_wmma_f32_16x16x32_f16(
                    /*neg_a=*/false, a, /*neg_b=*/false, b,
                    /*c_mod=*/(short)0, acc[nt],
                    /*reuse_a=*/false, /*reuse_b=*/false);
            }
        }
        __syncthreads();
    }

    // ---- store: C/D layout -> VGPR r holds rows r (lanes 0-15) / 8+r (16-31)
    if (fullM) {
        #pragma unroll
        for (int nt = 0; nt < 4; ++nt) {
            const int col = n0 + nt * 16 + l16;
            const float bv = bias ? bias[col] : 0.0f;
            #pragma unroll
            for (int r = 0; r < 8; ++r) {
                int gr = m0 + wave * 16 + g * 8 + r;
                float v = acc[nt][r] + bv;
                if (act) v = (v > 0.0f) ? v : expm1f(v);
                C[(size_t)gr * ldc + col] = v;
            }
        }
    } else {
        #pragma unroll
        for (int nt = 0; nt < 4; ++nt) {
            const int col = n0 + nt * 16 + l16;
            const float bv = bias ? bias[col] : 0.0f;
            #pragma unroll
            for (int r = 0; r < 8; ++r) {
                int gr = m0 + wave * 16 + g * 8 + r;
                if (gr < M) {
                    float v = acc[nt][r] + bv;
                    if (act) v = (v > 0.0f) ? v : expm1f(v);
                    C[(size_t)gr * ldc + col] = v;
                }
            }
        }
    }
}

// ---------------------------------------------------------------------------
// Per-node attention logits: als[n,h] = sum_c h[n,h,c]*a_s[h,c] (same for ald)
// ---------------------------------------------------------------------------
__global__ void k_logits(const float* __restrict__ hbuf,
                         const float* __restrict__ a_s,
                         const float* __restrict__ a_d,
                         float* __restrict__ als, float* __restrict__ ald,
                         int N, int H, int C)
{
    int idx = blockIdx.x * blockDim.x + threadIdx.x;
    if (idx >= N * H) return;
    int n = idx / H, h = idx % H;
    const float* row = hbuf + (size_t)n * H * C + (size_t)h * C;
    float s0 = 0.0f, s1 = 0.0f;
    for (int c = 0; c < C; ++c) {
        float hv = row[c];
        s0 += hv * a_s[h * C + c];
        s1 += hv * a_d[h * C + c];
    }
    als[idx] = s0;
    ald[idx] = s1;
}

__global__ void k_fill(float* __restrict__ p, float v, int n)
{
    int i = blockIdx.x * blockDim.x + threadIdx.x;
    if (i < n) p[i] = v;
}

// out[n, col0 + c] = bias[c]  (initialize conv slice with bias before scatter)
__global__ void k_init_bias(float* __restrict__ out, const float* __restrict__ bias,
                            int ldc, int col0, int N, int width)
{
    int i = blockIdx.x * blockDim.x + threadIdx.x;
    if (i >= N * width) return;
    int n = i / width, c = i % width;
    out[(size_t)n * ldc + col0 + c] = bias[c];
}

__global__ void k_elu(float* __restrict__ p, int n)
{
    int i = blockIdx.x * blockDim.x + threadIdx.x;
    if (i < n) {
        float v = p[i];
        p[i] = (v > 0.0f) ? v : expm1f(v);
    }
}

// ---------------------------------------------------------------------------
// Edge pass 1: segment max of leaky_relu(als[src]+ald[dst]) over dst
// one thread per (edge, head); self-loops are edges [E, E+N)
// ---------------------------------------------------------------------------
__global__ void k_edge_max(const int* __restrict__ src, const int* __restrict__ dst,
                           const int* __restrict__ typ,
                           const float* __restrict__ als, const float* __restrict__ ald,
                           float* __restrict__ m, int E, int N, int H, int conv)
{
    int idx = blockIdx.x * blockDim.x + threadIdx.x;
    int Eall = E + N;
    if (idx >= Eall * H) return;
    int e = idx / H, h = idx % H;
    int s, d;
    if (e < E) {
        if (conv < 4 && typ[e] != conv) return;
        s = src[e]; d = dst[e];
    } else {
        s = d = e - E;
    }
    float ev = als[s * H + h] + ald[d * H + h];
    ev = (ev >= 0.0f) ? ev : 0.2f * ev;
    atomicMaxF(&m[d * H + h], ev);
}

// Edge pass 2: denom[dst,h] += exp(e - m[dst,h])
__global__ void k_edge_sum(const int* __restrict__ src, const int* __restrict__ dst,
                           const int* __restrict__ typ,
                           const float* __restrict__ als, const float* __restrict__ ald,
                           const float* __restrict__ m, float* __restrict__ denom,
                           int E, int N, int H, int conv)
{
    int idx = blockIdx.x * blockDim.x + threadIdx.x;
    int Eall = E + N;
    if (idx >= Eall * H) return;
    int e = idx / H, h = idx % H;
    int s, d;
    if (e < E) {
        if (conv < 4 && typ[e] != conv) return;
        s = src[e]; d = dst[e];
    } else {
        s = d = e - E;
    }
    float ev = als[s * H + h] + ald[d * H + h];
    ev = (ev >= 0.0f) ? ev : 0.2f * ev;
    atomicAdd(&denom[d * H + h], expf(ev - m[d * H + h]));
}

// ---------------------------------------------------------------------------
// Edge pass 3: out[dst, col0 + f] += alpha(h) * hbuf[src, f]
// one wave32 per edge:
//   - lanes 0..H-1 compute per-head alpha ONCE, broadcast with __shfl
//   - lanes stream features 4-wide (b128 gathers + f32 global atomics)
// ---------------------------------------------------------------------------
__global__ __launch_bounds__(256) void k_edge_scatter(
    const int* __restrict__ src, const int* __restrict__ dst,
    const int* __restrict__ typ,
    const float* __restrict__ als, const float* __restrict__ ald,
    const float* __restrict__ m, const float* __restrict__ denom,
    const float* __restrict__ hbuf, float* __restrict__ out,
    int ldc, int col0, int E, int N, int H, int C, int conv)
{
    int wid  = (blockIdx.x * blockDim.x + threadIdx.x) >> 5;
    int lane = threadIdx.x & 31;
    int Eall = E + N;
    if (wid >= Eall) return;
    int s, d;
    if (wid < E) {
        if (conv < 4 && typ[wid] != conv) return;   // uniform across wave
        s = src[wid]; d = dst[wid];
    } else {
        s = d = wid - E;
    }
    const int HC = H * C;
    const float* hrow = hbuf + (size_t)s * HC;
    float* orow = out + (size_t)d * ldc + col0;
    __builtin_prefetch(hrow, 0, 0);                 // global_prefetch_b8 (L2-resident row)

    // per-head alpha computed once (lanes 0..H-1), broadcast to the wave
    float alpha_h = 0.0f;
    if (lane < H) {
        float ev = als[s * H + lane] + ald[d * H + lane];
        ev = (ev >= 0.0f) ? ev : 0.2f * ev;
        alpha_h = expf(ev - m[d * H + lane]) / denom[d * H + lane];
    }

    // 4 features per lane per iteration; C % 32 == 0 so a float4 never
    // crosses a head boundary and f0/C is exact per lane.
    for (int f0 = lane * 4; f0 < HC; f0 += 128) {
        float4 hv = *(const float4*)(hrow + f0);
        float alpha = __shfl(alpha_h, f0 / C);
        atomicAdd(&orow[f0 + 0], hv.x * alpha);
        atomicAdd(&orow[f0 + 1], hv.y * alpha);
        atomicAdd(&orow[f0 + 2], hv.z * alpha);
        atomicAdd(&orow[f0 + 3], hv.w * alpha);
    }
}

// ---------------------------------------------------------------------------
// Host-side driver
// ---------------------------------------------------------------------------
static inline void gemm_launch(const float* A, const float* B, const float* bias,
                               float* C, int M, int Ncols, int K,
                               int lda, int ldb, int ldc, int act, hipStream_t st)
{
    dim3 grid((M + 127) / 128, Ncols / 64);
    k_gemm<<<grid, 256, 0, st>>>(A, B, bias, C, M, Ncols, K, lda, ldb, ldc, act);
}

extern "C" void kernel_launch(void* const* d_in, const int* in_sizes, int n_in,
                              void* d_out, int out_size, void* d_ws, size_t ws_size,
                              hipStream_t stream)
{
    const float* x     = (const float*)d_in[0];   // [N,256]
    const int*   esrc  = (const int*)  d_in[1];   // [E]
    const int*   edst  = (const int*)  d_in[2];
    const int*   etype = (const int*)  d_in[3];
    const float* W1    = (const float*)d_in[4];   // [5,256,256]
    const float* as1   = (const float*)d_in[5];   // [5,8,32]
    const float* ad1   = (const float*)d_in[6];
    const float* b1    = (const float*)d_in[7];   // [5,256]
    const float* f1w   = (const float*)d_in[8];   // [1280,256]
    const float* f1b   = (const float*)d_in[9];   // [256]
    const float* W2    = (const float*)d_in[10];  // [5,256,64]
    const float* as2   = (const float*)d_in[11];  // [5,1,64]
    const float* ad2   = (const float*)d_in[12];
    const float* b2    = (const float*)d_in[13];  // [5,64]
    const float* f2w   = (const float*)d_in[14];  // [320,64]
    const float* f2b   = (const float*)d_in[15];  // [64]
    float* out = (float*)d_out;                   // [N,64]

    const int F = 256;
    const int N = in_sizes[0] / F;                // 20000
    const int E = in_sizes[1];                    // 640000
    const int Eall = E + N;

    // workspace layout (floats)
    float* hcat1 = (float*)d_ws;                        // N*1280
    float* h1    = hcat1 + (size_t)N * 1280;            // N*256
    float* htmp  = h1    + (size_t)N * 256;             // N*256 (per-conv h)
    float* hcat2 = htmp  + (size_t)N * 256;             // N*320
    float* als   = hcat2 + (size_t)N * 320;             // N*8 (max H)
    float* ald   = als   + (size_t)N * 8;
    float* mbuf  = ald   + (size_t)N * 8;
    float* dbuf  = mbuf  + (size_t)N * 8;
    (void)ws_size; (void)n_in; (void)out_size;

    const int TB = 256;
    auto nb = [](int n, int b) { return (n + b - 1) / b; };

    // ======================= Layer 1 (H=8, C=32, HC=256) ====================
    {
        const int H = 8, C = 32, HC = 256;
        for (int i = 0; i < 5; ++i) {
            // h_i = x @ W1[i]    [N,256]
            gemm_launch(x, W1 + (size_t)i * F * HC, nullptr, htmp,
                        N, HC, F, F, HC, HC, /*act=*/0, stream);
            // logits
            k_logits<<<nb(N * H, TB), TB, 0, stream>>>(
                htmp, as1 + (size_t)i * H * C, ad1 + (size_t)i * H * C,
                als, ald, N, H, C);
            // softmax state
            k_fill<<<nb(N * H, TB), TB, 0, stream>>>(mbuf, NEGV, N * H);
            k_fill<<<nb(N * H, TB), TB, 0, stream>>>(dbuf, 0.0f, N * H);
            // out slice <- bias
            k_init_bias<<<nb(N * HC, TB), TB, 0, stream>>>(
                hcat1, b1 + (size_t)i * HC, 1280, i * HC, N, HC);
            // edge passes
            k_edge_max<<<nb(Eall * H, TB), TB, 0, stream>>>(
                esrc, edst, etype, als, ald, mbuf, E, N, H, i);
            k_edge_sum<<<nb(Eall * H, TB), TB, 0, stream>>>(
                esrc, edst, etype, als, ald, mbuf, dbuf, E, N, H, i);
            k_edge_scatter<<<nb(Eall * 32, TB), TB, 0, stream>>>(
                esrc, edst, etype, als, ald, mbuf, dbuf,
                htmp, hcat1, 1280, i * HC, E, N, H, C, i);
        }
        // elu over concatenated conv outputs
        k_elu<<<nb(N * 1280, TB), TB, 0, stream>>>(hcat1, N * 1280);
        // h1 = elu(hcat1 @ fus1_w + fus1_b)   [N,256]
        gemm_launch(hcat1, f1w, f1b, h1, N, 256, 1280, 1280, 256, 256,
                    /*act=*/1, stream);
    }

    // ======================= Layer 2 (H=1, C=64, HC=64) =====================
    {
        const int H = 1, C = 64, HC = 64;
        for (int i = 0; i < 5; ++i) {
            // h_i = h1 @ W2[i]   [N,64]
            gemm_launch(h1, W2 + (size_t)i * 256 * HC, nullptr, htmp,
                        N, HC, 256, 256, HC, HC, /*act=*/0, stream);
            k_logits<<<nb(N * H, TB), TB, 0, stream>>>(
                htmp, as2 + (size_t)i * HC, ad2 + (size_t)i * HC,
                als, ald, N, H, C);
            k_fill<<<nb(N * H, TB), TB, 0, stream>>>(mbuf, NEGV, N * H);
            k_fill<<<nb(N * H, TB), TB, 0, stream>>>(dbuf, 0.0f, N * H);
            k_init_bias<<<nb(N * HC, TB), TB, 0, stream>>>(
                hcat2, b2 + (size_t)i * HC, 320, i * HC, N, HC);
            k_edge_max<<<nb(Eall * H, TB), TB, 0, stream>>>(
                esrc, edst, etype, als, ald, mbuf, E, N, H, i);
            k_edge_sum<<<nb(Eall * H, TB), TB, 0, stream>>>(
                esrc, edst, etype, als, ald, mbuf, dbuf, E, N, H, i);
            k_edge_scatter<<<nb(Eall * 32, TB), TB, 0, stream>>>(
                esrc, edst, etype, als, ald, mbuf, dbuf,
                htmp, hcat2, 320, i * HC, E, N, H, C, i);
        }
        k_elu<<<nb(N * 320, TB), TB, 0, stream>>>(hcat2, N * 320);
        // out = elu(hcat2 @ fus2_w + fus2_b)   [N,64]
        gemm_launch(hcat2, f2w, f2b, out, N, 64, 320, 320, 64, 64,
                    /*act=*/1, stream);
    }
}
